// LocalAdaptiveMamba_42210938585782
// MI455X (gfx1250) — matrix operand order repaired and checked by reading, not compiled, not run on hardware
//
#include <hip/hip_runtime.h>

// ---------------------------------------------------------------------------
// LocalAdaptiveMamba for gfx1250 (MI455X): WMMA-bf16 GEMM pipeline
//   - 128x64 block tiles, 8 waves, v_wmma_f32_16x16x32_bf16
//   - double-buffered LDS pipeline: 1 barrier / k-step, async B prefetch
//     via global_load_async_to_lds_b128 (ASYNCcnt) overlapped with WMMA
//   - native packed bf16 converts (v_cvt_pk_bf16_f32) for A staging
// ---------------------------------------------------------------------------

typedef __attribute__((ext_vector_type(16))) __bf16 v16bf;
typedef __attribute__((ext_vector_type(8)))  __bf16 v8bf;
typedef __attribute__((ext_vector_type(2)))  __bf16 v2bf;
typedef __attribute__((ext_vector_type(8)))  float  v8f;

static constexpr int HPIX = 48;
static constexpr int NPIX = 48 * 48;        // 2304
static constexpr int NSEQ = 2 * NPIX;       // 4608 sequences
static constexpr int CDIM = 256;
static constexpr int DIN  = 512;
static constexpr int NST  = 16;

// ---- workspace layout (static part; dynamic part sized by runtime K) ------
static constexpr size_t HDR_OFF   = 0;                       // int k, K, T
static constexpr size_t W1_OFF    = 256;                     // bf16 [4][1024][256]
static constexpr size_t W1_BYTES  = 4ull * 1024 * 256 * 2;
static constexpr size_t WDT_OFF   = W1_OFF + W1_BYTES;       // bf16 [4][512][512]
static constexpr size_t WDT_BYTES = 4ull * 512 * 512 * 2;
static constexpr size_t WBC_OFF   = WDT_OFF + WDT_BYTES;     // bf16 [4][32][512]
static constexpr size_t WBC_BYTES = 4ull * 32 * 512 * 2;
static constexpr size_t OW_OFF    = WBC_OFF + WBC_BYTES;     // bf16 [4][256][512]
static constexpr size_t OW_BYTES  = 4ull * 256 * 512 * 2;
static constexpr size_t GW_OFF    = OW_OFF + OW_BYTES;       // bf16 [4][64][256]
static constexpr size_t GW_BYTES  = 4ull * 64 * 256 * 2;
static constexpr size_t DYN_OFF   = ((GW_OFF + GW_BYTES + 255) / 256) * 256;

// dynamic buffers (fp32), ids:
// 0: P0   (T x 256)   ping-pong x / xh / match
// 1: P1   (T x 256)
// 2: U    (T x 512)
// 3: Z    (T x 512)
// 4: DT   (T x 512)   dt, later overwritten in-place with y*silu(z)
// 5: BC   (T x 32)
// 6: GEOM (T x 64)
__device__ inline float* bufP(char* ws, int id) {
    const int* hdr = (const int*)ws;
    size_t T = (size_t)hdr[2];
    static const int widths[7] = {256, 256, 512, 512, 512, 32, 64};
    char* p = ws + DYN_OFF;
    for (int i = 0; i < id; ++i) p += T * (size_t)widths[i] * 4;
    return (float*)p;
}

__device__ inline unsigned short f2bf(float f) {           // RNE via native cvt
    return __builtin_bit_cast(unsigned short, (__bf16)f);
}
__device__ inline unsigned pkbf(float x, float y) {        // v_cvt_pk_bf16_f32
    v2bf v = { (__bf16)x, (__bf16)y };
    return __builtin_bit_cast(unsigned, v);
}

// ---------------------------------------------------------------------------
// 0) spans reduction -> k, K=k*k, T=4608*K
// ---------------------------------------------------------------------------
__global__ __launch_bounds__(256) void spans_k(const int* sx, const int* sy, int* hdr) {
    int tid = threadIdx.x;
    int s = 0;
    for (int i = tid; i < NPIX; i += 256) s += sx[i] + sy[i];
    for (int o = 16; o > 0; o >>= 1) s += __shfl_down(s, o, 32);
    __shared__ int red[8];
    if ((tid & 31) == 0) red[tid >> 5] = s;
    __syncthreads();
    if (tid == 0) {
        int tot = 0;
        for (int w = 0; w < 8; ++w) tot += red[w];
        int k = tot / NSEQ;                 // floor of mean (non-negative)
        if (k > 15) k = 15;
        if (k < 3)  k = 3;
        hdr[0] = k; hdr[1] = k * k; hdr[2] = NSEQ * k * k;
    }
}

// ---------------------------------------------------------------------------
// 1) weight preparation (bf16 conversion + algebraic folding)
// ---------------------------------------------------------------------------
__global__ __launch_bounds__(256) void prep_w1(const float* ipw, const float* nw,
                                               unsigned short* W1) {
    size_t idx = (size_t)blockIdx.x * 256 + threadIdx.x;   // 4*1024*256
    int d = (int)(idx / (1024 * 256));
    int c = (int)(idx % 256);
    W1[idx] = f2bf(ipw[idx] * nw[d * 256 + c]);
}

__global__ __launch_bounds__(256) void prep_wdt(const float* dtw, const float* xpw,
                                                unsigned short* WDT) {
    size_t idx = (size_t)blockIdx.x * 256 + threadIdx.x;   // 4*512*512
    int d = (int)(idx / (512 * 512));
    int rem = (int)(idx % (512 * 512));
    int o = rem / 512, c = rem % 512;
    const float* dw = dtw + ((size_t)d * 512 + o) * 16;
    const float* xp = xpw + (size_t)d * 48 * 512 + c;
    float acc = 0.f;
    for (int r = 0; r < 16; ++r) acc += dw[r] * xp[(size_t)r * 512];
    WDT[idx] = f2bf(acc);
}

__global__ __launch_bounds__(256) void prep_wbc(const float* xpw, unsigned short* WBC) {
    size_t idx = (size_t)blockIdx.x * 256 + threadIdx.x;   // 4*32*512
    int d = (int)(idx / (32 * 512));
    int rem = (int)(idx % (32 * 512));
    int n = rem / 512, c = rem % 512;
    WBC[idx] = f2bf(xpw[(size_t)d * 48 * 512 + (size_t)(16 + n) * 512 + c]);
}

__global__ __launch_bounds__(256) void cvt_k(const float* src, unsigned short* dst, int n) {
    int idx = blockIdx.x * 256 + threadIdx.x;
    if (idx < n) dst[idx] = f2bf(src[idx]);
}

// ---------------------------------------------------------------------------
// 2) unfold: build token stream x[m*K + t][c]
// ---------------------------------------------------------------------------
__global__ __launch_bounds__(256) void unfold_k(const float* f0, const float* f1, char* ws) {
    const int* hdr = (const int*)ws;
    int k = hdr[0], K = hdr[1], pad = hdr[0] >> 1;
    float* P0 = bufP(ws, 0);
    int m = blockIdx.x;
    int img = m / NPIX, pix = m % NPIX;
    int i = pix / HPIX, j = pix % HPIX;
    const float* src = img ? f1 : f0;
    int c = threadIdx.x;
    for (int t = 0; t < K; ++t) {
        int di = t / k, dj = t % k;
        int si = i + di - pad, sj = j + dj - pad;
        float v = 0.f;
        if (si >= 0 && si < HPIX && sj >= 0 && sj < HPIX)
            v = src[(size_t)c * NPIX + si * HPIX + sj];
        P0[((size_t)m * K + t) * CDIM + c] = v;
    }
}

// ---------------------------------------------------------------------------
// 3) RMS scale: xh = x * rsqrt(mean(x^2)+eps)   (norm_w folded into W1)
//    wave-per-token, barrier-free
// ---------------------------------------------------------------------------
__global__ __launch_bounds__(256) void rms_k(char* ws, int cur) {
    const int* hdr = (const int*)ws;
    int K = hdr[1];
    float* Xa = bufP(ws, cur);
    float* Xb = bufP(ws, cur ^ 1);
    int m = blockIdx.x;
    int wave = threadIdx.x >> 5, lane = threadIdx.x & 31;
    for (int t = wave; t < K; t += 8) {
        size_t base = ((size_t)m * K + t) * CDIM;
        float x[8];
        float v = 0.f;
        for (int j = 0; j < 8; ++j) {
            x[j] = Xa[base + j * 32 + lane];
            v += x[j] * x[j];
        }
        for (int o = 16; o > 0; o >>= 1) v += __shfl_down(v, o, 32);
        v = __shfl(v, 0, 32);
        float s = rsqrtf(v * (1.f / 256.f) + 1e-5f);
        for (int j = 0; j < 8; ++j) Xb[base + j * 32 + lane] = x[j] * s;
    }
}

// ---------------------------------------------------------------------------
// 4) WMMA GEMM: Y[T x Nout] = X[T x KIN] @ W^T (+epilogue)
//    128x64 tile / block, 8 waves; double-buffered, 1 barrier per k-step
//    EPI: 0 = store, 1 = +bias, softplus, 2 = +residual, 3 = +bias
// ---------------------------------------------------------------------------
__device__ inline v16bf load_afrag(const __bf16* At, int mOff, int lane) {
    int half = lane >> 4, mr = lane & 15;
    const __bf16* row = At + (mOff + mr) * 40;
    v8bf lo = *(const v8bf*)(row + half * 8);        // K 0..7 (lane<16) / 8..15
    v8bf hi = *(const v8bf*)(row + 16 + half * 8);   // K 16..23 / 24..31
    return __builtin_shufflevector(lo, hi, 0,1,2,3,4,5,6,7,8,9,10,11,12,13,14,15);
}
__device__ inline v16bf load_bfrag(const __bf16* Bt, int nOff, int lane) {
    int half = lane >> 4, nc = lane & 15;
    const __bf16* row = Bt + (nOff + nc) * 40 + half * 8;   // K 0..7 / 8..15, then +16
    v8bf lo = *(const v8bf*)(row);
    v8bf hi = *(const v8bf*)(row + 16);
    return __builtin_shufflevector(lo, hi, 0,1,2,3,4,5,6,7,8,9,10,11,12,13,14,15);
}

struct ARegs { float4 a, b, c, d; };

template <int KIN, int EPI>
__global__ __launch_bounds__(256) void gemm_k(char* ws, int xid, int yid,
                                              const unsigned short* Wt,
                                              const float* bias, int rid, int Nout) {
    const int* hdr = (const int*)ws;
    int T = hdr[2];
    int mTiles = T >> 7;                        // T is always a multiple of 128
    float* X = bufP(ws, xid);
    float* Y = bufP(ws, yid);
    const float* R = (EPI == 2) ? bufP(ws, rid) : nullptr;

    __shared__ __bf16 At[2][128 * 40];
    __shared__ __bf16 Bt[2][64 * 40];
    constexpr int BT_BYTES = 64 * 40 * 2;       // per-buffer B bytes

    int tid = threadIdx.x;
    int lane = tid & 31;
    int mOff = (tid >> 5) * 16;                 // wave's 16-row strip
    int nBase = blockIdx.y * 64;
    int ar = tid >> 1, ac = (tid & 1) * 16;     // A staging: 16 floats / thread
    int br = tid >> 2, bc = (tid & 3) * 8;      // B staging: 16 bytes / thread
    bool bvalid = (nBase + br) < Nout;
    unsigned ldsB0 = (unsigned)(size_t)&Bt[0][br * 40 + bc];
    const unsigned short* wrow = Wt + (size_t)(nBase + br) * KIN + bc;

    // zero-fill B rows beyond Nout once (both buffers); async never writes them
    if (Nout < 64) {
        for (int i = Nout * 40 + tid; i < 64 * 40; i += 256) {
            Bt[0][i] = (__bf16)0.f;
            Bt[1][i] = (__bf16)0.f;
        }
    }

    constexpr int NSTEPS = KIN / 32;

    for (int tile = blockIdx.x; tile < mTiles; tile += gridDim.x) {
        size_t mBase = (size_t)tile * 128;
        const float* xrow = X + (mBase + ar) * KIN + ac;
        v8f acc0 = {}, acc1 = {}, acc2 = {}, acc3 = {};

        // ---- prologue: stage k-step 0 into buffer 0 ----
        if (bvalid) {
            asm volatile("global_load_async_to_lds_b128 %0, %1, off"
                         :: "v"(ldsB0), "v"(wrow) : "memory");
        }
        {
            ARegs r;
            r.a = *(const float4*)(xrow);
            r.b = *(const float4*)(xrow + 4);
            r.c = *(const float4*)(xrow + 8);
            r.d = *(const float4*)(xrow + 12);
            uint4 p0, p1;
            p0.x = pkbf(r.a.x, r.a.y); p0.y = pkbf(r.a.z, r.a.w);
            p0.z = pkbf(r.b.x, r.b.y); p0.w = pkbf(r.b.z, r.b.w);
            p1.x = pkbf(r.c.x, r.c.y); p1.y = pkbf(r.c.z, r.c.w);
            p1.z = pkbf(r.d.x, r.d.y); p1.w = pkbf(r.d.z, r.d.w);
            *(uint4*)&At[0][ar * 40 + ac]     = p0;
            *(uint4*)&At[0][ar * 40 + ac + 8] = p1;
        }
        asm volatile("s_wait_asynccnt 0x0" ::: "memory");
        __syncthreads();

        // ---- pipelined main loop: 1 barrier per k-step ----
        for (int i = 0; i < NSTEPS; ++i) {
            int p = i & 1;
            bool more = (i + 1) < NSTEPS;
            int kn = (i + 1) * 32;

            // issue next B (async -> LDS) and next A (global -> regs) early
            ARegs r;
            if (more) {
                if (bvalid) {
                    asm volatile("global_load_async_to_lds_b128 %0, %1, off"
                                 :: "v"(ldsB0 + (unsigned)((p ^ 1) * BT_BYTES)),
                                    "v"(wrow + kn) : "memory");
                }
                const float* src = xrow + kn;
                r.a = *(const float4*)(src);
                r.b = *(const float4*)(src + 4);
                r.c = *(const float4*)(src + 8);
                r.d = *(const float4*)(src + 12);
            }

            // compute current step from buffer p
            v16bf a  = load_afrag(At[p], mOff, lane);
            v16bf b0 = load_bfrag(Bt[p], 0,  lane);
            v16bf b1 = load_bfrag(Bt[p], 16, lane);
            v16bf b2 = load_bfrag(Bt[p], 32, lane);
            v16bf b3 = load_bfrag(Bt[p], 48, lane);
            acc0 = __builtin_amdgcn_wmma_f32_16x16x32_bf16(false, a, false, b0,
                                                           (short)0, acc0, false, false);
            acc1 = __builtin_amdgcn_wmma_f32_16x16x32_bf16(false, a, false, b1,
                                                           (short)0, acc1, false, false);
            acc2 = __builtin_amdgcn_wmma_f32_16x16x32_bf16(false, a, false, b2,
                                                           (short)0, acc2, false, false);
            acc3 = __builtin_amdgcn_wmma_f32_16x16x32_bf16(false, a, false, b3,
                                                           (short)0, acc3, false, false);

            // commit next A tile (cvt + ds_store) into the other buffer
            if (more) {
                uint4 p0, p1;
                p0.x = pkbf(r.a.x, r.a.y); p0.y = pkbf(r.a.z, r.a.w);
                p0.z = pkbf(r.b.x, r.b.y); p0.w = pkbf(r.b.z, r.b.w);
                p1.x = pkbf(r.c.x, r.c.y); p1.y = pkbf(r.c.z, r.c.w);
                p1.z = pkbf(r.d.x, r.d.y); p1.w = pkbf(r.d.z, r.d.w);
                *(uint4*)&At[p ^ 1][ar * 40 + ac]     = p0;
                *(uint4*)&At[p ^ 1][ar * 40 + ac + 8] = p1;
            }
            asm volatile("s_wait_asynccnt 0x0" ::: "memory");
            __syncthreads();
        }

        // ---- epilogue ----
        int half = lane >> 4, nl = lane & 15;
        for (int nt = 0; nt < 4; ++nt) {
            v8f acc = (nt == 0) ? acc0 : (nt == 1) ? acc1 : (nt == 2) ? acc2 : acc3;
            int nG = nBase + nt * 16 + nl;
            if (nG < Nout) {
                for (int j = 0; j < 8; ++j) {
                    size_t tok = mBase + mOff + j + half * 8;
                    float v = acc[j];
                    if (EPI == 1) { v += bias[nG]; v = (v > 20.f) ? v : log1pf(expf(v)); }
                    else if (EPI == 2) { v += R[tok * (size_t)Nout + nG]; }
                    else if (EPI == 3) { v += bias[nG]; }
                    Y[tok * (size_t)Nout + nG] = v;
                }
            }
        }
    }
}

// ---------------------------------------------------------------------------
// 5) causal depthwise conv (DCONV=4) + SiLU, in place on U
// ---------------------------------------------------------------------------
__global__ __launch_bounds__(256) void conv_k(char* ws, const float* cw, const float* cb) {
    const int* hdr = (const int*)ws;
    int K = hdr[1];
    float* U = bufP(ws, 2);
    int gid = blockIdx.x * 256 + threadIdx.x;
    int m = gid >> 9, ch = gid & 511;
    float w0 = cw[ch * 4 + 0], w1 = cw[ch * 4 + 1];
    float w2 = cw[ch * 4 + 2], w3 = cw[ch * 4 + 3];
    float b = cb[ch];
    float u0 = 0.f, u1 = 0.f, u2 = 0.f;
    size_t base = (size_t)m * K * DIN + ch;
    for (int t = 0; t < K; ++t) {
        float u3 = U[base + (size_t)t * DIN];
        float y = b + u0 * w0 + u1 * w1 + u2 * w2 + u3 * w3;
        y = y / (1.f + expf(-y));                  // SiLU
        U[base + (size_t)t * DIN] = y;
        u0 = u1; u1 = u2; u2 = u3;
    }
}

// ---------------------------------------------------------------------------
// 6) selective-scan; writes y*silu(z) in place into DT buffer
// ---------------------------------------------------------------------------
__global__ __launch_bounds__(256) void scan_k(char* ws, const float* alog, const float* dp) {
    const int* hdr = (const int*)ws;
    int K = hdr[1];
    float* U  = bufP(ws, 2);
    float* Z  = bufP(ws, 3);
    float* DT = bufP(ws, 4);
    float* BC = bufP(ws, 5);
    int m  = blockIdx.x >> 1;
    int ch = ((blockIdx.x & 1) << 8) + threadIdx.x;
    float a[NST], s[NST];
    for (int n = 0; n < NST; ++n) { a[n] = -expf(alog[ch * NST + n]); s[n] = 0.f; }
    float D = dp[ch];
    __shared__ float sbc[32];
    for (int t = 0; t < K; ++t) {
        size_t idx = (size_t)m * K + t;
        if (threadIdx.x < 32) sbc[threadIdx.x] = BC[idx * 32 + threadIdx.x];
        __syncthreads();
        float dt = DT[idx * DIN + ch];
        float u  = U[idx * DIN + ch];
        float du = dt * u;
        float y  = u * D;
        for (int n = 0; n < NST; ++n) {
            s[n] = s[n] * expf(dt * a[n]) + du * sbc[n];
            y += s[n] * sbc[16 + n];
        }
        float z  = Z[idx * DIN + ch];
        float sz = z / (1.f + expf(-z));
        DT[idx * DIN + ch] = y * sz;
        __syncthreads();
    }
}

// ---------------------------------------------------------------------------
// 7) mean over K + NCHW transpose into d_out
// ---------------------------------------------------------------------------
__global__ __launch_bounds__(256) void final_k(char* ws, int cur, float* out) {
    const int* hdr = (const int*)ws;
    int K = hdr[1];
    float* M = bufP(ws, cur);
    float* G = bufP(ws, 6);
    int m = blockIdx.x, c = threadIdx.x;
    int img = m / NPIX, pix = m % NPIX;
    float inv = 1.f / (float)K;
    float accM = 0.f;
    for (int t = 0; t < K; ++t) accM += M[((size_t)m * K + t) * CDIM + c];
    out[(size_t)img * CDIM * NPIX + (size_t)c * NPIX + pix] = accM * inv;
    if (c < 64) {
        float accG = 0.f;
        for (int t = 0; t < K; ++t) accG += G[((size_t)m * K + t) * 64 + c];
        out[2ull * CDIM * NPIX + (size_t)img * 64 * NPIX + (size_t)c * NPIX + pix] = accG * inv;
    }
}

// ---------------------------------------------------------------------------
// host driver
// ---------------------------------------------------------------------------
extern "C" void kernel_launch(void* const* d_in, const int* in_sizes, int n_in,
                              void* d_out, int out_size, void* d_ws, size_t ws_size,
                              hipStream_t stream) {
    const float* f0     = (const float*)d_in[0];
    const float* f1     = (const float*)d_in[1];
    const int*   sx0    = (const int*)d_in[4];
    const int*   sy0    = (const int*)d_in[5];
    const float* norm_w = (const float*)d_in[8];
    const float* ipw    = (const float*)d_in[9];
    const float* conv_w = (const float*)d_in[10];
    const float* conv_b = (const float*)d_in[11];
    const float* xpw    = (const float*)d_in[12];
    const float* dtw    = (const float*)d_in[13];
    const float* dtb    = (const float*)d_in[14];
    const float* alog   = (const float*)d_in[15];
    const float* dp     = (const float*)d_in[16];
    const float* ow     = (const float*)d_in[17];
    const float* gw     = (const float*)d_in[18];
    const float* gb     = (const float*)d_in[19];

    char* ws = (char*)d_ws;
    unsigned short* W1  = (unsigned short*)(ws + W1_OFF);
    unsigned short* WDT = (unsigned short*)(ws + WDT_OFF);
    unsigned short* WBC = (unsigned short*)(ws + WBC_OFF);
    unsigned short* OW  = (unsigned short*)(ws + OW_OFF);
    unsigned short* GW  = (unsigned short*)(ws + GW_OFF);

    spans_k<<<1, 256, 0, stream>>>(sx0, sy0, (int*)ws);
    prep_w1<<<4096, 256, 0, stream>>>(ipw, norm_w, W1);
    prep_wdt<<<4096, 256, 0, stream>>>(dtw, xpw, WDT);
    prep_wbc<<<256, 256, 0, stream>>>(xpw, WBC);
    cvt_k<<<2048, 256, 0, stream>>>(ow, OW, 4 * 256 * 512);
    cvt_k<<<256, 256, 0, stream>>>(gw, GW, 4 * 64 * 256);
    unfold_k<<<NSEQ, 256, 0, stream>>>(f0, f1, ws);

    int cur = 0;
    for (int d = 0; d < 4; ++d) {
        rms_k<<<NSEQ, 256, 0, stream>>>(ws, cur);
        // u = xh @ W1[:512]^T ; z = xh @ W1[512:]^T
        gemm_k<256, 0><<<dim3(1024, 8), 256, 0, stream>>>(
            ws, cur ^ 1, 2, W1 + (size_t)d * 1024 * 256, nullptr, 0, 512);
        gemm_k<256, 0><<<dim3(1024, 8), 256, 0, stream>>>(
            ws, cur ^ 1, 3, W1 + (size_t)d * 1024 * 256 + 512 * 256, nullptr, 0, 512);
        conv_k<<<9216, 256, 0, stream>>>(ws, conv_w + (size_t)d * 512 * 4,
                                         conv_b + (size_t)d * 512);
        // dt = softplus(u @ Wdt^T + dtb)
        gemm_k<512, 1><<<dim3(1024, 8), 256, 0, stream>>>(
            ws, 2, 4, WDT + (size_t)d * 512 * 512, dtb + (size_t)d * 512, 0, 512);
        // [B|C] = u @ Wbc^T
        gemm_k<512, 0><<<dim3(1024, 1), 256, 0, stream>>>(
            ws, 2, 5, WBC + (size_t)d * 32 * 512, nullptr, 0, 32);
        scan_k<<<9216, 256, 0, stream>>>(ws, alog + (size_t)d * 512 * 16,
                                         dp + (size_t)d * 512);
        // match = x + yz @ ow^T
        gemm_k<512, 2><<<dim3(1024, 4), 256, 0, stream>>>(
            ws, 4, cur ^ 1, OW + (size_t)d * 256 * 512, nullptr, cur, 256);
        cur ^= 1;
    }
    // geom = match @ gw^T + gb   (last depth only)
    gemm_k<256, 3><<<dim3(1024, 1), 256, 0, stream>>>(
        ws, cur, 6, GW + 3ull * 64 * 256, gb + 3 * 64, 0, 64);
    final_k<<<NSEQ, 256, 0, stream>>>(ws, cur, (float*)d_out);
}
